// TriangleSelfAttentionV49_46961172414801
// MI455X (gfx1250) — matrix operand order repaired and checked by reading, not compiled, not run on hardware
//
#include <hip/hip_runtime.h>
#include <hip/hip_bf16.h>

// ---------------------------------------------------------------------------
// TriangleSelfAttention (N=256, C=128, H=4, Ch=32) for gfx1250 (CDNA5).
// All GEMM-like math runs through v_wmma_f32_16x16x32_bf16; attention tile
// staging uses double-buffered global_load_async_to_lds_b128 (ASYNCcnt) with
// partial s_wait_asynccnt for software pipelining.
// Pipeline: prep(weights->bf16) -> proj(q,k,v,gate) -> zero attn acc ->
// attention core (two-pass online softmax over n, WMMA logits + WMMA AV,
// j-split w/ float atomics) -> out-projection + gate + residual + LayerNorm.
// Workspace: q_bf16[16M] k_bf16[16M] v_bf16[16M] gate_bf16[16M]
//            attn_f32[32M] wqkvg_bf16[128K] wo_bf16[32K]
// ---------------------------------------------------------------------------

typedef __attribute__((ext_vector_type(16))) __bf16 bf16x16;
typedef __attribute__((ext_vector_type(8)))  float  floatx8;
typedef unsigned short ushort_t;

#define NSEQ   256
#define CDIM   128
#define CH     32
#define QSCALE 0.17677669529663687f   // 1/sqrt(32)

// Native conversions: backend emits v_cvt_pk_bf16_f32 / bf16 cvt ops.
__device__ inline __bf16 f2bf(float f) { return (__bf16)f; }
__device__ inline ushort_t f2bfu(float f) {
  return __builtin_bit_cast(ushort_t, (__bf16)f);
}
__device__ inline float bfu2f(ushort_t s) {
  unsigned u = ((unsigned)s) << 16;
  return __builtin_bit_cast(float, u);
}

__device__ inline floatx8 wmma_bf16(bf16x16 a, bf16x16 b, floatx8 c) {
  return __builtin_amdgcn_wmma_f32_16x16x32_bf16(false, a, false, b, (short)0, c,
                                                 false, false);
}

// Async 16-byte global -> LDS copy (per lane), tracked by ASYNCcnt.
__device__ inline void async_copy16(void* lds_dst, const void* gsrc) {
  asm volatile("global_load_async_to_lds_b128 %0, %1, off"
               :
               : "v"((unsigned)(unsigned long long)lds_dst), "v"(gsrc)
               : "memory");
}
__device__ inline void wait_async0() {
  asm volatile("s_wait_asynccnt 0x0" ::: "memory");
}
// Each staging batch = 5 async instructions per wave; <=5 outstanding means
// the previous batch has fully landed (double-buffer pipelining).
__device__ inline void wait_async5() {
  asm volatile("s_wait_asynccnt 0x5" ::: "memory");
}

// A-fragment from bf16 memory: two contiguous 8-element runs (callers use contiguous k = 16h+s order)
__device__ inline bf16x16 fragA16(const ushort_t* p0, const ushort_t* p1) {
  union { bf16x16 v; uint4 q[2]; } r;
  r.q[0] = *(const uint4*)p0;
  r.q[1] = *(const uint4*)p1;
  return r.v;
}
// B-fragment from bf16 memory: 16 contiguous elements (contiguous k = 16h+s order)
__device__ inline bf16x16 fragB16(const ushort_t* p) {
  union { bf16x16 v; uint4 q[2]; } r;
  r.q[0] = *(const uint4*)p;
  r.q[1] = *(const uint4*)(p + 8);
  return r.v;
}
// A-fragment from two 8-float global runs, convert to bf16 (native cvt)
__device__ inline bf16x16 gfragA(const float* __restrict__ p0,
                                 const float* __restrict__ p1) {
  bf16x16 v;
#pragma unroll
  for (int e = 0; e < 8; ++e) { v[e] = f2bf(p0[e]); v[e + 8] = f2bf(p1[e]); }
  return v;
}
__device__ inline floatx8 zero8() { floatx8 z = {0,0,0,0,0,0,0,0}; return z; }

// ---------------------------------------------------------------------------
// Kernel W: one-time weight conversion f32 -> bf16 (QSCALE folded into q_w).
// ---------------------------------------------------------------------------
__global__ __launch_bounds__(256) void tsa_prep_kernel(
    const float* __restrict__ qw, const float* __restrict__ kw,
    const float* __restrict__ vw, const float* __restrict__ gw,
    const float* __restrict__ ow, ushort_t* __restrict__ wcat,
    ushort_t* __restrict__ wo_bf) {
  const int idx = blockIdx.x * 256 + threadIdx.x;   // 0 .. 81919
  const int m = idx >> 14, r = idx & 16383;
  if (m == 0)      wcat[r]             = f2bfu(qw[r] * QSCALE);
  else if (m == 1) wcat[16384 + r]     = f2bfu(kw[r]);
  else if (m == 2) wcat[2 * 16384 + r] = f2bfu(vw[r]);
  else if (m == 3) wcat[3 * 16384 + r] = f2bfu(gw[r]);
  else             wo_bf[r]            = f2bfu(ow[r]);
}

// ---------------------------------------------------------------------------
// Kernel A: fused q/k/v/gate projection.  X(65536x128) @ Wt(128x512) in bf16
// WMMA, f32 accumulate.  64 rows/WG, 8 waves; wave w owns column stripe
// w*16..w*16+15 of each of the 4 weight matrices (t = matrix).
// ---------------------------------------------------------------------------
__global__ __launch_bounds__(256) void tsa_proj_kernel(
    const float* __restrict__ pair, const ushort_t* __restrict__ wcat,
    const float* __restrict__ qb, const float* __restrict__ kb,
    const float* __restrict__ vb, const float* __restrict__ gb,
    ushort_t* __restrict__ q_bf, ushort_t* __restrict__ k_bf,
    ushort_t* __restrict__ v_bf, ushort_t* __restrict__ g_bf) {
  const int tid = threadIdx.x;
  const int w = tid >> 5, L = tid & 31;
  const int lm = L & 15, lh = L >> 4;
  const int row0 = blockIdx.x * 64;

  const float* Bia[4] = {qb, kb, vb, gb};

  floatx8 acc[4][4];
#pragma unroll
  for (int mt = 0; mt < 4; ++mt)
#pragma unroll
    for (int t = 0; t < 4; ++t) acc[mt][t] = zero8();

#pragma unroll
  for (int ks = 0; ks < 4; ++ks) {            // K = 128 in 4 steps of 32
    bf16x16 afr[4];
#pragma unroll
    for (int mt = 0; mt < 4; ++mt) {
      const int row = row0 + mt * 16 + lm;
      const int c0 = ks * 32 + lh * 16;
      afr[mt] = gfragA(pair + row * CDIM + c0, pair + row * CDIM + c0 + 8);
    }
#pragma unroll
    for (int t = 0; t < 4; ++t) {             // t selects weight matrix
      const int wo = w * 16 + lm;             // column within that matrix
      const int c0b = ks * 32 + lh * 16;
      bf16x16 bfr = fragB16(wcat + t * 16384 + wo * CDIM + c0b);
#pragma unroll
      for (int mt = 0; mt < 4; ++mt) acc[mt][t] = wmma_bf16(afr[mt], bfr, acc[mt][t]);
    }
  }
  // epilogue: bias, sigmoid for gate, bf16 store
#pragma unroll
  for (int t = 0; t < 4; ++t) {
    const int wo = w * 16 + lm;
    const float bias = (t == 0) ? Bia[0][wo] * QSCALE : Bia[t][wo];
#pragma unroll
    for (int mt = 0; mt < 4; ++mt) {
#pragma unroll
      for (int r = 0; r < 8; ++r) {
        const int row = row0 + mt * 16 + r + lh * 8;
        float val = acc[mt][t][r] + bias;
        const int idx = row * CDIM + wo;
        if (t == 0)      q_bf[idx] = f2bfu(val);
        else if (t == 1) k_bf[idx] = f2bfu(val);
        else if (t == 2) v_bf[idx] = f2bfu(val);
        else             g_bf[idx] = f2bfu(1.0f / (1.0f + __expf(-val)));
      }
    }
  }
}

// ---------------------------------------------------------------------------
// Kernel Z: zero the f32 attention accumulator (needed for j-split atomics).
// ---------------------------------------------------------------------------
__global__ __launch_bounds__(256) void tsa_zero_kernel(float* __restrict__ p) {
  const int base = (blockIdx.x * 256 + threadIdx.x) * 16;
  float4 z = {0.f, 0.f, 0.f, 0.f};
#pragma unroll
  for (int i = 0; i < 4; ++i) *(float4*)(p + base + i * 4) = z;
}

// ---------------------------------------------------------------------------
// Kernel B: attention core.  Grid = 4(h) * 16(i-tile of 16) * 4(j-chunk of 64).
// Two passes over n (softmax axis), NB=4 n per iteration, 8 waves:
//   wave -> (jt = w&3 : j-tile within chunk, part = w>>2 : n pair half, nn = 2*part+qq).
// Q/K staged with double-buffered async-to-LDS copies; V prefetched into
// registers and stored c-major into a double-buffered LDS tile.
// ---------------------------------------------------------------------------
__global__ __launch_bounds__(256) void tsa_attn_kernel(
    const ushort_t* __restrict__ q_bf, const ushort_t* __restrict__ k_bf,
    const ushort_t* __restrict__ v_bf, const unsigned char* __restrict__ mask,
    float* __restrict__ attn_f) {
  __shared__ __align__(16) ushort_t lds_q[2][4][16][32];   //  8 KB
  __shared__ __align__(16) ushort_t lds_k[2][4][64][32];   // 32 KB
  __shared__ __align__(16) ushort_t lds_v[2][4][32][64];   // 32 KB (c-major)
  __shared__ __align__(16) ushort_t lds_a[4][16][64];      //  8 KB
  __shared__ __align__(16) float    lds_m[2][16][64];      //  8 KB
  __shared__ __align__(16) float    lds_s[2][16][64];      //  8 KB

  const int tid = threadIdx.x;
  const int w = tid >> 5, L = tid & 31;
  const int lm = L & 15, lh = L >> 4;
  const int bx = blockIdx.x;
  const int h = bx >> 6, itile = (bx >> 2) & 15, jc = bx & 3;
  const int i0 = itile * 16, jbase = jc * 64;
  const int jt = w & 3, part = w >> 2;
  const int jl_lane = jt * 16 + lm;          // local j column for this lane
  const bool mvalid = mask[jbase + jl_lane] != 0;
  const int hoff = h * CH;

  // loop-invariant staging indices / global offsets (n0 adds n0*CDIM)
  const int qf = tid * 8;
  const int q_nn = qf >> 9, q_il = (qf & 511) >> 5, q_c = qf & 31;
  const int qoff = ((i0 + q_il) * NSEQ + q_nn) * CDIM + hoff + q_c;
  int k_nn[4], k_jl[4], k_c[4], koff[4];
#pragma unroll
  for (int rep = 0; rep < 4; ++rep) {
    const int f = (tid + rep * 256) * 8;
    k_nn[rep] = f >> 11;
    k_jl[rep] = (f & 2047) >> 5;
    k_c[rep] = f & 31;
    koff[rep] = ((jbase + k_jl[rep]) * NSEQ + k_nn[rep]) * CDIM + hoff + k_c[rep];
  }

  float mreg[8], sreg[8];
#pragma unroll
  for (int r = 0; r < 8; ++r) { mreg[r] = -3.0e38f; sreg[r] = 0.f; }

  // issue one Q/K staging batch (5 async instructions per wave)
  auto issueQK = [&](int n0, int buf) {
    async_copy16(&lds_q[buf][q_nn][q_il][q_c], q_bf + qoff + n0 * CDIM);
#pragma unroll
    for (int rep = 0; rep < 4; ++rep)
      async_copy16(&lds_k[buf][k_nn[rep]][k_jl[rep]][k_c[rep]],
                   k_bf + koff[rep] + n0 * CDIM);
  };

  // ---------------- pass 1: running max / sum over n ----------------
  issueQK(0, 0);
  for (int it = 0; it < 64; ++it) {
    const int buf = it & 1;
    if (it < 63) { issueQK((it + 1) * 4, buf ^ 1); wait_async5(); }
    else wait_async0();
    __syncthreads();
#pragma unroll
    for (int qq = 0; qq < 2; ++qq) {
      const int nn = part * 2 + qq;
      bf16x16 a = fragA16(&lds_q[buf][nn][lm][lh * 16],
                          &lds_q[buf][nn][lm][lh * 16 + 8]);
      bf16x16 b = fragB16(&lds_k[buf][nn][jl_lane][lh * 16]);
      floatx8 d = wmma_bf16(a, b, zero8());
#pragma unroll
      for (int r = 0; r < 8; ++r) {
        float l = mvalid ? d[r] : -1.0e9f;
        float nm = fmaxf(mreg[r], l);
        sreg[r] = sreg[r] * __expf(mreg[r] - nm) + __expf(l - nm);
        mreg[r] = nm;
      }
    }
    __syncthreads();
  }

  // pipeline prologue for pass 2 (overlaps with the m/s combine below)
  issueQK(0, 0);
  uint4 vpre[4];
#pragma unroll
  for (int rep = 0; rep < 4; ++rep)
    vpre[rep] = *(const uint4*)&v_bf[koff[rep]];

  // combine the two n-half partial (m,s) states per (i,j)
#pragma unroll
  for (int r = 0; r < 8; ++r) {
    const int il = r + lh * 8;
    lds_m[part][il][jl_lane] = mreg[r];
    lds_s[part][il][jl_lane] = sreg[r];
  }
  __syncthreads();
  float Mr[8], invS[8];
#pragma unroll
  for (int r = 0; r < 8; ++r) {
    const int il = r + lh * 8;
    const float ma = lds_m[0][il][jl_lane], mb = lds_m[1][il][jl_lane];
    const float M = fmaxf(ma, mb);
    const float S = lds_s[0][il][jl_lane] * __expf(ma - M) +
                    lds_s[1][il][jl_lane] * __expf(mb - M);
    Mr[r] = M; invS[r] = 1.0f / S;
  }
  __syncthreads();

  // ---------------- pass 2: normalized attn + AV contraction ----------------
  for (int it = 0; it < 64; ++it) {
    const int buf = it & 1;
    const int n0 = it * 4;
    if (it < 63) issueQK(n0 + 4, buf ^ 1);
    // store this iteration's prefetched V (c-major) into its LDS buffer
#pragma unroll
    for (int rep = 0; rep < 4; ++rep) {
      union { uint4 u4; ushort_t s[8]; } t;
      t.u4 = vpre[rep];
#pragma unroll
      for (int e = 0; e < 8; ++e)
        lds_v[buf][k_nn[rep]][k_c[rep] + e][k_jl[rep]] = t.s[e];
    }
    if (it < 63) { // prefetch next V into registers
#pragma unroll
      for (int rep = 0; rep < 4; ++rep)
        vpre[rep] = *(const uint4*)&v_bf[koff[rep] + (n0 + 4) * CDIM];
      wait_async5();
    } else wait_async0();
    __syncthreads();
    // phase A: recompute logits, normalize, stage bf16 attn in LDS
#pragma unroll
    for (int qq = 0; qq < 2; ++qq) {
      const int nn = part * 2 + qq;
      bf16x16 a = fragA16(&lds_q[buf][nn][lm][lh * 16],
                          &lds_q[buf][nn][lm][lh * 16 + 8]);
      bf16x16 b = fragB16(&lds_k[buf][nn][jl_lane][lh * 16]);
      floatx8 d = wmma_bf16(a, b, zero8());
#pragma unroll
      for (int r = 0; r < 8; ++r) {
        float l = mvalid ? d[r] : -1.0e9f;
        float av = __expf(l - Mr[r]) * invS[r];
        lds_a[nn][r + lh * 8][jl_lane] = f2bfu(av);
      }
    }
    __syncthreads();
    // phase B: Out_n = A_n @ V_n  (one (n, c-tile) task per wave)
    {
      const int nn = w >> 1, ct = w & 1;
      floatx8 acc = zero8();
#pragma unroll
      for (int ksj = 0; ksj < 2; ++ksj) {      // K = 64 j in 2 steps of 32
        bf16x16 a = fragA16(&lds_a[nn][lm][ksj * 32 + lh * 16],
                            &lds_a[nn][lm][ksj * 32 + lh * 16 + 8]);
        bf16x16 b = fragB16(&lds_v[buf][nn][ct * 16 + lm][ksj * 32 + lh * 16]);
        acc = wmma_bf16(a, b, acc);
      }
      const int n = n0 + nn;
#pragma unroll
      for (int r = 0; r < 8; ++r) {
        const int il = r + lh * 8, c = ct * 16 + lm;
        unsafeAtomicAdd(&attn_f[(((i0 + il) * NSEQ) + n) * CDIM + hoff + c], acc[r]);
      }
    }
    __syncthreads();
  }
}

// ---------------------------------------------------------------------------
// Kernel C: gated out-projection + bias + residual + LayerNorm.
// ---------------------------------------------------------------------------
__global__ __launch_bounds__(256) void tsa_outln_kernel(
    const float* __restrict__ attn_f, const ushort_t* __restrict__ g_bf,
    const ushort_t* __restrict__ wo_bf, const float* __restrict__ ob,
    const float* __restrict__ pair, const float* __restrict__ lng,
    const float* __restrict__ lnb, float* __restrict__ out) {
  __shared__ __align__(16) float sres[64][CDIM];   // 32 KB
  __shared__ float ps[64][4], ps2[64][4];

  const int tid = threadIdx.x;
  const int w = tid >> 5, L = tid & 31;
  const int lm = L & 15, lh = L >> 4;
  const int row0 = blockIdx.x * 64;

  floatx8 acc[4];
#pragma unroll
  for (int mt = 0; mt < 4; ++mt) acc[mt] = zero8();

#pragma unroll
  for (int ks = 0; ks < 4; ++ks) {
    bf16x16 afr[4];
#pragma unroll
    for (int mt = 0; mt < 4; ++mt) {
      const int row = row0 + mt * 16 + lm;
      const int c0 = ks * 32 + lh * 16;
      bf16x16 v;
#pragma unroll
      for (int e = 0; e < 8; ++e) {
        const int ca = c0 + e, cb = c0 + 8 + e;
        v[e]     = f2bf(attn_f[row * CDIM + ca] * bfu2f(g_bf[row * CDIM + ca]));
        v[e + 8] = f2bf(attn_f[row * CDIM + cb] * bfu2f(g_bf[row * CDIM + cb]));
      }
      afr[mt] = v;
    }
    const int o = w * 16 + lm;
    const int c0b = ks * 32 + lh * 16;
    bf16x16 bfr = fragB16(wo_bf + o * CDIM + c0b);
#pragma unroll
    for (int mt = 0; mt < 4; ++mt) acc[mt] = wmma_bf16(afr[mt], bfr, acc[mt]);
  }
  // bias + residual into LDS for the cross-lane LN reduction
#pragma unroll
  for (int mt = 0; mt < 4; ++mt) {
#pragma unroll
    for (int r = 0; r < 8; ++r) {
      const int rl = mt * 16 + r + lh * 8;
      const int o = w * 16 + lm;
      sres[rl][o] = acc[mt][r] + ob[o] + pair[(row0 + rl) * CDIM + o];
    }
  }
  __syncthreads();
  // LayerNorm: 4 threads per row, 32 channels each
  {
    const int rl = tid >> 2, seg = tid & 3;
    float s = 0.f, s2 = 0.f;
#pragma unroll
    for (int c = seg * 32; c < seg * 32 + 32; ++c) {
      const float v = sres[rl][c];
      s += v; s2 += v * v;
    }
    ps[rl][seg] = s; ps2[rl][seg] = s2;
  }
  __syncthreads();
  {
    const int rl = tid >> 2, seg = tid & 3;
    const float s  = ps[rl][0] + ps[rl][1] + ps[rl][2] + ps[rl][3];
    const float s2 = ps2[rl][0] + ps2[rl][1] + ps2[rl][2] + ps2[rl][3];
    const float mu = s * (1.0f / CDIM);
    const float var = s2 * (1.0f / CDIM) - mu * mu;
    const float inv = rsqrtf(var + 1e-5f);
    const int row = row0 + rl;
#pragma unroll
    for (int c = seg * 32; c < seg * 32 + 32; ++c)
      out[row * CDIM + c] = (sres[rl][c] - mu) * inv * lng[c] + lnb[c];
  }
}

// ---------------------------------------------------------------------------
extern "C" void kernel_launch(void* const* d_in, const int* in_sizes, int n_in,
                              void* d_out, int out_size, void* d_ws, size_t ws_size,
                              hipStream_t stream) {
  const float* pair = (const float*)d_in[0];
  const unsigned char* mask = (const unsigned char*)d_in[1];
  const float* q_w = (const float*)d_in[2];
  const float* q_b = (const float*)d_in[3];
  const float* k_w = (const float*)d_in[4];
  const float* k_b = (const float*)d_in[5];
  const float* v_w = (const float*)d_in[6];
  const float* v_b = (const float*)d_in[7];
  const float* g_w = (const float*)d_in[8];
  const float* g_b = (const float*)d_in[9];
  const float* o_w = (const float*)d_in[10];
  const float* o_b = (const float*)d_in[11];
  const float* ln_g = (const float*)d_in[12];
  const float* ln_b = (const float*)d_in[13];
  float* out = (float*)d_out;

  const size_t NROW = (size_t)NSEQ * NSEQ;          // 65536
  const size_t NEL = NROW * CDIM;                   // 8,388,608
  ushort_t* q_bf = (ushort_t*)d_ws;
  ushort_t* k_bf = q_bf + NEL;
  ushort_t* v_bf = k_bf + NEL;
  ushort_t* g_bf = v_bf + NEL;
  float* attn_f = (float*)(g_bf + NEL);             // byte offset 64 MB, 32 MB f32
  ushort_t* wcat = (ushort_t*)(attn_f + NEL);       // 4 x 16384 bf16
  ushort_t* wo_bf = wcat + 4 * 16384;               // 16384 bf16

  (void)in_sizes; (void)n_in; (void)out_size; (void)ws_size;

  tsa_prep_kernel<<<320, 256, 0, stream>>>(q_w, k_w, v_w, g_w, o_w, wcat, wo_bf);
  tsa_proj_kernel<<<1024, 256, 0, stream>>>(pair, wcat, q_b, k_b, v_b, g_b,
                                            q_bf, k_bf, v_bf, g_bf);
  tsa_zero_kernel<<<(int)(NEL / (256 * 16)), 256, 0, stream>>>(attn_f);
  tsa_attn_kernel<<<256, 256, 0, stream>>>(q_bf, k_bf, v_bf, mask, attn_f);
  tsa_outln_kernel<<<1024, 256, 0, stream>>>(attn_f, g_bf, wo_bf, o_b, pair,
                                             ln_g, ln_b, out);
}
